// CMAALayer_56014963475220
// MI455X (gfx1250) — matrix operand order repaired and checked by reading, not compiled, not run on hardware
//
#include <hip/hip_runtime.h>
#include <hip/hip_bf16.h>
#include <math.h>
#include <stdint.h>

// ---- MI455X (gfx1250) fused CMAA layer ----
// 86 GFLOP of GEMM -> v_wmma_f32_16x16x32_f16 (f32 accumulate), ~0.4GB HBM traffic.
// M-tile = 32 rows/block so each B-fragment feeds 2 WMMAs (halves L2 B traffic).
// A tiles staged via global_load_async_to_lds_b128 (ASYNCcnt) + s_wait_asynccnt.

typedef __attribute__((ext_vector_type(16))) _Float16 v16h;
typedef __attribute__((ext_vector_type(8)))  _Float16 v8h;
typedef __attribute__((ext_vector_type(8)))  float    v8f;

#define NBATCH 32768
#define ROWS   (NBATCH * 4)          // 131072 token rows
#define DDIM   256
#define HEADS  8
#define DHEAD  32
#define MT     32                    // rows per block (2 WMMA M-tiles)
#define SCALE  0.17677669529663687f  // 1/sqrt(32)
#define Y_ELEMS ((size_t)ROWS * DDIM)     // 33554432 floats of y, then attn

// workspace byte offsets
#define XH_OFF  ((size_t)0)                          // f16 X, reused as f16 O
#define QH_OFF  ((size_t)ROWS * DDIM * 2)            // 64MB each
#define KH_OFF  (QH_OFF * 2)
#define VH_OFF  (QH_OFF * 3)
#define WC_OFF  (QH_OFF * 4)                         // Wcat f16: 1024x256
#define WO_OFF  (WC_OFF + (size_t)1024 * 256 * 2)    // Wo f16: 256x256

struct h4 { _Float16 a, b, c, d; };

// Async-copy 64B/thread (global -> LDS), tracked with ASYNCcnt.
__device__ __forceinline__ void async_tile64(const _Float16* g, _Float16* l) {
  uint32_t lds = (uint32_t)(uintptr_t)l;     // low 32 bits = LDS byte offset
  uint64_t ga  = (uint64_t)(uintptr_t)g;
  asm volatile(
      "global_load_async_to_lds_b128 %0, %1, off\n\t"
      "global_load_async_to_lds_b128 %0, %1, off offset:16\n\t"
      "global_load_async_to_lds_b128 %0, %1, off offset:32\n\t"
      "global_load_async_to_lds_b128 %0, %1, off offset:48"
      :: "v"(lds), "v"(ga) : "memory");
  asm volatile("s_wait_asynccnt 0x0" ::: "memory");
}

// ---------------- conversion kernels ----------------
__global__ void k_cvt_tokens(const float* __restrict__ x, _Float16* __restrict__ xh, int n4) {
  int t = blockIdx.x * blockDim.x + threadIdx.x;
  if (t >= n4) return;
  float4 v = ((const float4*)x)[t];
  h4 o = { (_Float16)v.x, (_Float16)v.y, (_Float16)v.z, (_Float16)v.w };
  ((h4*)xh)[t] = o;
}

__global__ void k_cvt_w(const float* __restrict__ wq, const float* __restrict__ wk,
                        const float* __restrict__ wv, const float* __restrict__ wr,
                        const float* __restrict__ wo,
                        _Float16* __restrict__ wcat, _Float16* __restrict__ woh) {
  int t = blockIdx.x * blockDim.x + threadIdx.x;     // 0..81919
  if (t >= 5 * 65536 / 4) return;
  int idx   = t * 4;
  int which = idx >> 16;                             // 0..4 : Wq Wk Wv Wr Wo
  int off   = idx & 65535;
  const float* srcs[5] = { wq, wk, wv, wr, wo };
  float4 v = *(const float4*)(srcs[which] + off);
  h4 o = { (_Float16)v.x, (_Float16)v.y, (_Float16)v.z, (_Float16)v.w };
  _Float16* dst = (which < 4) ? (wcat + (size_t)which * 65536 + off) : (woh + off);
  *(h4*)dst = o;
}

// ---------------- fused Q/K/V/R projection (WMMA) ----------------
// Block = 256 thr = 8 wave32, M-tile = 32 rows, N = 1024 (Wq|Wk|Wv|Wr stacked).
// Wave w owns N-tiles {w+8j}: j=0,1 -> Q ; 2,3 -> K ; 4,5 -> V ; 6,7 -> R
// so sigmoid(R)*V gating happens on in-register accumulators.
__global__ __launch_bounds__(256) void k_proj(const _Float16* __restrict__ xh,
                                              const _Float16* __restrict__ wcat,
                                              _Float16* __restrict__ qh,
                                              _Float16* __restrict__ kh,
                                              _Float16* __restrict__ vh) {
  __shared__ __align__(32) _Float16 As[MT * 256];    // 16KB A tile
  const int tid     = threadIdx.x;
  const int rowbase = blockIdx.x * MT;
  async_tile64(xh + (size_t)rowbase * DDIM + tid * 32, &As[tid * 32]);
  __syncthreads();

  const int w = tid >> 5, lane = tid & 31;
  const int half = lane >> 4, lr = lane & 15;
  v8f zero = {};
  v8f acc[2][8];
#pragma unroll
  for (int m = 0; m < 2; ++m)
#pragma unroll
    for (int j = 0; j < 8; ++j) acc[m][j] = zero;

  union { v16h v; v8h h[2]; } af[2];
#pragma unroll
  for (int ks = 0; ks < 8; ++ks) {
    const int kb = ks * 32;
#pragma unroll
    for (int m = 0; m < 2; ++m) {
      const _Float16* ap = &As[(m * 16 + lr) * 256 + kb + half * 16];
      af[m].h[0] = *(const v8h*)ap;                   // K = kb+0..7   (or 16..23)
      af[m].h[1] = *(const v8h*)(ap + 8);             // K = kb+8..15  (or 24..31)
    }
#pragma unroll
    for (int j = 0; j < 8; ++j) {
      const int nt = w + 8 * j;
      const _Float16* bp = wcat + ((size_t)(nt * 16 + lr)) * 256 + kb + half * 16;
      v16h bfrag = *(const v16h*)bp;
      __builtin_prefetch(bp + 32, 0, 3);              // next K-step's B
      acc[0][j] = __builtin_amdgcn_wmma_f32_16x16x32_f16(
          false, af[0].v, false, bfrag, (short)0, acc[0][j], false, false);
      acc[1][j] = __builtin_amdgcn_wmma_f32_16x16x32_f16(
          false, af[1].v, false, bfrag, (short)0, acc[1][j], false, false);
    }
  }

#pragma unroll
  for (int m = 0; m < 2; ++m) {
    const int rb = rowbase + m * 16;
#pragma unroll
    for (int j = 0; j < 2; ++j) {
      const int cb = (w + 8 * j) * 16;
#pragma unroll
      for (int v = 0; v < 8; ++v) {
        const size_t ro = (size_t)(rb + v + 8 * half) * DDIM + cb + lr;
        qh[ro] = (_Float16)acc[m][j][v];
        kh[ro] = (_Float16)acc[m][2 + j][v];
        float rg = acc[m][6 + j][v];
        float g  = 1.0f / (1.0f + __expf(-rg));
        vh[ro] = (_Float16)(g * acc[m][4 + j][v]);
      }
    }
  }
}

// ---------------- attention over 3 neighbors (VALU, tiny) ----------------
__global__ void k_attn(const _Float16* __restrict__ qh, const _Float16* __restrict__ kh,
                       const _Float16* __restrict__ vh, _Float16* __restrict__ oh,
                       float* __restrict__ attn_out) {
  int tid = blockIdx.x * blockDim.x + threadIdx.x;
  if (tid >= ROWS * HEADS) return;
  const int r = tid >> 3, h = tid & 7;
  const int b = r >> 2, i = r & 3;

  float qf[DHEAD];
  const _Float16* q = qh + (size_t)r * DDIM + h * DHEAD;
#pragma unroll
  for (int c = 0; c < 4; ++c) {
    v8h t = ((const v8h*)q)[c];
#pragma unroll
    for (int j = 0; j < 8; ++j) qf[c * 8 + j] = (float)t[j];
  }

  int   rl[3];
  float s[3];
#pragma unroll
  for (int l = 0; l < 3; ++l) {
    const int n = (l >= i) ? (l + 1) : l;             // NEIGH[i][l]
    rl[l] = b * 4 + n;
    const _Float16* kp = kh + (size_t)rl[l] * DDIM + h * DHEAD;
    float d = 0.f;
#pragma unroll
    for (int c = 0; c < 4; ++c) {
      v8h t = ((const v8h*)kp)[c];
#pragma unroll
      for (int j = 0; j < 8; ++j) d += qf[c * 8 + j] * (float)t[j];
    }
    s[l] = d * SCALE;
  }
  float m  = fmaxf(s[0], fmaxf(s[1], s[2]));
  float e0 = __expf(s[0] - m), e1 = __expf(s[1] - m), e2 = __expf(s[2] - m);
  float inv = 1.0f / (e0 + e1 + e2);
  float a[3] = { e0 * inv, e1 * inv, e2 * inv };

  float o[DHEAD];
#pragma unroll
  for (int j = 0; j < DHEAD; ++j) o[j] = 0.f;
#pragma unroll
  for (int l = 0; l < 3; ++l) {
    const _Float16* vp = vh + (size_t)rl[l] * DDIM + h * DHEAD;
#pragma unroll
    for (int c = 0; c < 4; ++c) {
      v8h t = ((const v8h*)vp)[c];
#pragma unroll
      for (int j = 0; j < 8; ++j) o[c * 8 + j] += a[l] * (float)t[j];
    }
  }
  _Float16* op = oh + (size_t)r * DDIM + h * DHEAD;
#pragma unroll
  for (int j = 0; j < DHEAD; ++j) op[j] = (_Float16)o[j];

  float* ao = attn_out + ((size_t)r * HEADS + h) * 3;
  ao[0] = a[0]; ao[1] = a[1]; ao[2] = a[2];
}

// ---------------- O @ Wo.T + residual + LayerNorm (WMMA) ----------------
__global__ __launch_bounds__(256) void k_out(const _Float16* __restrict__ oh,
                                             const _Float16* __restrict__ woh,
                                             const float* __restrict__ tokens,
                                             const float* __restrict__ ln_g,
                                             const float* __restrict__ ln_b,
                                             float* __restrict__ y) {
  __shared__ __align__(32) _Float16 As[MT * 256];    // 16KB
  __shared__ float Ys[MT * 256];                     // 32KB staging
  __shared__ float Mu[MT], Rst[MT];
  const int tid = threadIdx.x;
  const int rowbase = blockIdx.x * MT;
  async_tile64(oh + (size_t)rowbase * DDIM + tid * 32, &As[tid * 32]);
  __syncthreads();

  const int w = tid >> 5, lane = tid & 31;
  const int half = lane >> 4, lr = lane & 15;
  v8f zero = {};
  v8f acc[2][2];
#pragma unroll
  for (int m = 0; m < 2; ++m) { acc[m][0] = zero; acc[m][1] = zero; }
  union { v16h v; v8h h[2]; } af[2];
#pragma unroll
  for (int ks = 0; ks < 8; ++ks) {
    const int kb = ks * 32;
#pragma unroll
    for (int m = 0; m < 2; ++m) {
      const _Float16* ap = &As[(m * 16 + lr) * 256 + kb + half * 16];
      af[m].h[0] = *(const v8h*)ap;
      af[m].h[1] = *(const v8h*)(ap + 8);
    }
#pragma unroll
    for (int j = 0; j < 2; ++j) {
      const int nt = w + 8 * j;
      const _Float16* bp = woh + ((size_t)(nt * 16 + lr)) * 256 + kb + half * 16;
      v16h bfrag = *(const v16h*)bp;
      acc[0][j] = __builtin_amdgcn_wmma_f32_16x16x32_f16(
          false, af[0].v, false, bfrag, (short)0, acc[0][j], false, false);
      acc[1][j] = __builtin_amdgcn_wmma_f32_16x16x32_f16(
          false, af[1].v, false, bfrag, (short)0, acc[1][j], false, false);
    }
  }
#pragma unroll
  for (int m = 0; m < 2; ++m)
#pragma unroll
    for (int j = 0; j < 2; ++j) {
      const int nb = (w + 8 * j) * 16;
#pragma unroll
      for (int v = 0; v < 8; ++v)
        Ys[(m * 16 + v + 8 * half) * 256 + nb + lr] = acc[m][j][v];
    }
  __syncthreads();

  {   // residual add: thread handles 32 contiguous cols of one row
    const int row = tid >> 3, c0 = (tid & 7) * 32;
    const float* tp = tokens + (size_t)(rowbase + row) * DDIM + c0;
    float* yp = &Ys[row * 256 + c0];
#pragma unroll
    for (int j = 0; j < 32; ++j) yp[j] += tp[j];
  }
  __syncthreads();
  if (tid < MT) {
    float s = 0.f, ss = 0.f;
    for (int j = 0; j < 256; ++j) { float v = Ys[tid * 256 + j]; s += v; ss += v * v; }
    float mu  = s * (1.0f / 256.0f);
    float var = ss * (1.0f / 256.0f) - mu * mu;
    Mu[tid]  = mu;
    Rst[tid] = rsqrtf(var + 1e-5f);
  }
  __syncthreads();
  {
    const int row = tid >> 3, c0 = (tid & 7) * 32;
    const float mu = Mu[row], rs = Rst[row];
    float* dst = y + (size_t)(rowbase + row) * DDIM + c0;
#pragma unroll
    for (int j = 0; j < 32; ++j)
      dst[j] = (Ys[row * 256 + c0 + j] - mu) * rs * ln_g[c0 + j] + ln_b[c0 + j];
  }
}

extern "C" void kernel_launch(void* const* d_in, const int* in_sizes, int n_in,
                              void* d_out, int out_size, void* d_ws, size_t ws_size,
                              hipStream_t stream) {
  const float* tokens = (const float*)d_in[0];
  const float* wq = (const float*)d_in[1];
  const float* wk = (const float*)d_in[2];
  const float* wv = (const float*)d_in[3];
  const float* wr = (const float*)d_in[4];
  const float* wo = (const float*)d_in[5];
  const float* lg = (const float*)d_in[6];
  const float* lb = (const float*)d_in[7];

  char* ws = (char*)d_ws;
  _Float16* xh   = (_Float16*)(ws + XH_OFF);   // also reused as O (f16)
  _Float16* qh   = (_Float16*)(ws + QH_OFF);
  _Float16* kh   = (_Float16*)(ws + KH_OFF);
  _Float16* vh   = (_Float16*)(ws + VH_OFF);
  _Float16* wcat = (_Float16*)(ws + WC_OFF);
  _Float16* woh  = (_Float16*)(ws + WO_OFF);

  float* y    = (float*)d_out;
  float* attn = (float*)d_out + Y_ELEMS;

  const int n4 = ROWS * DDIM / 4;
  k_cvt_tokens<<<(n4 + 255) / 256, 256, 0, stream>>>(tokens, xh, n4);
  k_cvt_w<<<(5 * 65536 / 4 + 255) / 256, 256, 0, stream>>>(wq, wk, wv, wr, wo, wcat, woh);
  k_proj<<<ROWS / MT, 256, 0, stream>>>(xh, wcat, qh, kh, vh);
  k_attn<<<ROWS * HEADS / 256, 256, 0, stream>>>(qh, kh, vh, xh, attn);
  k_out<<<ROWS / MT, 256, 0, stream>>>(xh, woh, tokens, lg, lb, y);
}
